// FastBigBirdSelfAttention_8830452760955
// MI455X (gfx1250) — matrix operand order repaired and checked by reading, not compiled, not run on hardware
//
#include <hip/hip_runtime.h>
#include <hip/hip_bf16.h>

// ---------------------------------------------------------------------------
// FastBigBird self-attention for MI455X (gfx1250, wave32, WMMA)
//   H=12 heads, D=64, HID=768, T=1536, FW=128 window, 8 global keys/head
// ---------------------------------------------------------------------------

typedef __bf16 bf16;
typedef __attribute__((ext_vector_type(16))) __bf16 v16bf;
typedef __attribute__((ext_vector_type(8)))  __bf16 v8bf;
typedef __attribute__((ext_vector_type(8)))  float  v8f;
typedef __attribute__((ext_vector_type(4)))  float  vf4;

#define T_LEN 1536
#define HID   768
#define NHEAD 12
#define HD    64
#define FW    128
#define UNW   144          // union local window (128 + 15, padded to 9 tiles)
#define NGLB  8
#define NPAD  160          // 144 local + 8 global + 8 zero-pad = 10 N tiles
#define TAU_F 32.0f
#define ALPHA 0.1f

__device__ __forceinline__ int iclamp(int v, int lo, int hi) {
    return v < lo ? lo : (v > hi ? hi : v);
}

// Combine two 8-element bf16 runs into one 16-element WMMA fragment.
__device__ __forceinline__ v16bf make_frag(v8bf lo, v8bf hi) {
    v16bf r;
#pragma unroll
    for (int e = 0; e < 8; ++e) { r[e] = lo[e]; r[e + 8] = hi[e]; }
    return r;
}

// Convert 8 consecutive floats to a packed v8bf (two b128 loads, one b128 store).
__device__ __forceinline__ void cvt8(const float* __restrict__ src,
                                     bf16* __restrict__ dst) {
    const vf4* p = (const vf4*)src;
    vf4 a = p[0], b = p[1];
    v8bf o;
#pragma unroll
    for (int e = 0; e < 4; ++e) { o[e] = (bf16)a[e]; o[e + 4] = (bf16)b[e]; }
    *(v8bf*)dst = o;
}

// ---------------------------------------------------------------------------
// 1) fp32 -> bf16 conversion (8 elements per thread, fully vectorized)
// ---------------------------------------------------------------------------
__global__ __launch_bounds__(256) void convert_bf16_kernel(
    const float* __restrict__ X, const float* __restrict__ Wq,
    const float* __restrict__ Wk, const float* __restrict__ Wv,
    bf16* __restrict__ Xb, bf16* __restrict__ Wqb,
    bf16* __restrict__ Wkb, bf16* __restrict__ Wvb) {
    int i = (blockIdx.x * 256 + threadIdx.x) * 8;
    if (i < T_LEN * HID) cvt8(X + i, Xb + i);
    if (i < HID * HID) {
        cvt8(Wq + i, Wqb + i);
        cvt8(Wk + i, Wkb + i);
        cvt8(Wv + i, Wvb + i);
    }
}

// ---------------------------------------------------------------------------
// 2) Projection GEMM: Out[h][t][d] = sum_i X[t][i] * W[o][i] + bias[o],
//    o = h*64+d.  One wave -> one 16x16 tile, K=768 in 24 steps of 32 (bf16).
//    A layout (16-bit 16x32): lane = M%16, K = (e&7) + (e>=8?16:0) + (lane>=16?8:0)
//    B layout (16-bit 32x16): lane%16 = N, K = (e&7) + (e>=8?16:0) + (lane>=16?8:0)
// ---------------------------------------------------------------------------
__global__ __launch_bounds__(256) void qkv_gemm_kernel(
    const bf16* __restrict__ X, const bf16* __restrict__ W,
    const float* __restrict__ bias, float* __restrict__ Out /*[H][T][D]*/) {
    const int lane = threadIdx.x & 31;
    const int wave = threadIdx.x >> 5;
    const int mt   = blockIdx.x;               // 0..95  (T/16)
    const int nt   = blockIdx.y * 8 + wave;    // 0..47  (HID/16)
    const int l16  = lane & 15;
    const int hiA  = (lane >> 4) ? 8 : 0;
    const int hiB  = (lane >> 4) ? 16 : 0;

    const int row = mt * 16 + l16;             // t index for A
    const int col = nt * 16 + l16;             // o index for B / D

    v8f acc = {};
    const bf16* xrow = X + row * HID;
    const bf16* wrow = W + col * HID;
    for (int kc = 0; kc < HID; kc += 32) {
        const bf16* ap = xrow + kc + hiA;
        const bf16* bp = wrow + kc + hiA;
        v16bf a = make_frag(*(const v8bf*)ap, *(const v8bf*)(ap + 16));
        v16bf b = make_frag(*(const v8bf*)bp, *(const v8bf*)(bp + 16));
        acc = __builtin_amdgcn_wmma_f32_16x16x32_bf16(
            false, a, false, b, (short)0, acc, false, false);
    }

    const int h = col >> 6, d = col & 63;
    const float bi = bias[col];
    const int rbase = mt * 16 + ((lane >> 4) ? 8 : 0);
#pragma unroll
    for (int r = 0; r < 8; ++r) {
        int t = rbase + r;
        Out[(h * T_LEN + t) * HD + d] = acc[r] + bi;
    }
}

// ---------------------------------------------------------------------------
// 3) Per-head key statistics -> 8 global key indices per head
//    s[t] = ||K[t]|| + 0.25*||K[t]-K[t-1]||, argmax per group of 384
// ---------------------------------------------------------------------------
__global__ __launch_bounds__(256) void stats_kernel(
    const float* __restrict__ Kh, int* __restrict__ gidx) {
    __shared__ float s[T_LEN];
    const int h = blockIdx.x;
    const float* Kp = Kh + h * T_LEN * HD;
    for (int t = threadIdx.x; t < T_LEN; t += 256) {
        const vf4* kt = (const vf4*)(Kp + t * HD);
        const vf4* km = (const vf4*)(Kp + (t > 0 ? t - 1 : t) * HD);
        float nk = 0.f, nd = 0.f;
#pragma unroll 4
        for (int j = 0; j < HD / 4; ++j) {
            vf4 a = kt[j], b = km[j];
#pragma unroll
            for (int e = 0; e < 4; ++e) {
                float df = a[e] - b[e];
                nk += a[e] * a[e];
                nd += df * df;
            }
        }
        s[t] = sqrtf(nk) + 0.25f * sqrtf(nd);
    }
    __syncthreads();
    if (threadIdx.x < 4) {
        const int g = threadIdx.x, base = g * (T_LEN / 4);
        float best = s[base];
        int bi = 0;
        for (int j = 1; j < T_LEN / 4; ++j) {
            float v = s[base + j];
            if (v > best) { best = v; bi = j; }
        }
        gidx[h * NGLB + g] = base + bi;
    }
    if (threadIdx.x == 4) {
        gidx[h * NGLB + 4] = 512;       // round(linspace(0,1535,4)[1])
        gidx[h * NGLB + 5] = 1023;      // round(linspace(0,1535,4)[2])
        gidx[h * NGLB + 6] = 0;
        gidx[h * NGLB + 7] = T_LEN - 1;
    }
}

// ---------------------------------------------------------------------------
// 4) Attention: one wave per (head, 16-query tile).
//    Union local window of 144 keys + 8 global keys, padded to 160 columns.
//    scores: 10 Ntiles x 2 Kchunks WMMA;  ctx: 4 Dtiles x 5 Kchunks WMMA.
// ---------------------------------------------------------------------------
__global__ __launch_bounds__(32) void attn_kernel(
    const float* __restrict__ Qh, const float* __restrict__ Kh,
    const float* __restrict__ Vh, const int* __restrict__ gidx,
    float* __restrict__ Out /*[T][HID]*/) {
    __shared__ bf16  qn[16 * HD];      // [m][d]      normalized queries
    __shared__ bf16  kn[NPAD * HD];    // [col][d]    normalized keys
    __shared__ bf16  vt[HD * NPAD];    // [d][col]    values (transposed)
    __shared__ float sc[16 * NPAD];    // [m][col]    raw scores
    __shared__ bf16  wt[16 * NPAD];    // [m][col]    softmax weights

    const int mt   = blockIdx.x;                 // query tile 0..95
    const int h    = blockIdx.y;                 // head
    const int lane = threadIdx.x;
    const int t0   = mt * 16;
    const int lo   = iclamp(t0 - FW / 2, 0, T_LEN - FW);  // union window start
    const int l16  = lane & 15;
    const int ha   = (lane >> 4) ? 8 : 0;
    const int hb   = (lane >> 4) ? 16 : 0;

    // ---- stage normalized Q (lanes 0..15, one query row each) ----
    if (lane < 16) {
        const vf4* qp = (const vf4*)(Qh + (h * T_LEN + t0 + lane) * HD);
        vf4 buf[16];
        float ss = 0.f;
#pragma unroll
        for (int j = 0; j < 16; ++j) {
            buf[j] = qp[j];
#pragma unroll
            for (int e = 0; e < 4; ++e) ss += buf[j][e] * buf[j][e];
        }
        float inv = 1.f / fmaxf(sqrtf(ss), 1e-6f);
#pragma unroll
        for (int j = 0; j < 8; ++j) {
            v8bf o;
#pragma unroll
            for (int e = 0; e < 4; ++e) {
                o[e]     = (bf16)(buf[2 * j][e] * inv);
                o[e + 4] = (bf16)(buf[2 * j + 1][e] * inv);
            }
            *(v8bf*)(qn + lane * HD + j * 8) = o;
        }
    }

    // ---- stage normalized K and transposed V (5 columns per lane) ----
    for (int c = lane; c < NPAD; c += 32) {
        int ki = -1;
        if (c < UNW) { int k = lo + c; if (k < T_LEN) ki = k; }
        else if (c < UNW + NGLB) ki = gidx[h * NGLB + (c - UNW)];
        if (ki >= 0) {
            const vf4* kp = (const vf4*)(Kh + (h * T_LEN + ki) * HD);
            const vf4* vp = (const vf4*)(Vh + (h * T_LEN + ki) * HD);
            vf4 kb[16];
            float ss = 0.f;
#pragma unroll
            for (int j = 0; j < 16; ++j) {
                kb[j] = kp[j];
#pragma unroll
                for (int e = 0; e < 4; ++e) ss += kb[j][e] * kb[j][e];
            }
            float inv = 1.f / fmaxf(sqrtf(ss), 1e-6f);
#pragma unroll
            for (int j = 0; j < 8; ++j) {
                v8bf o;
#pragma unroll
                for (int e = 0; e < 4; ++e) {
                    o[e]     = (bf16)(kb[2 * j][e] * inv);
                    o[e + 4] = (bf16)(kb[2 * j + 1][e] * inv);
                }
                *(v8bf*)(kn + c * HD + j * 8) = o;
            }
#pragma unroll
            for (int j = 0; j < 16; ++j) {
                vf4 v = vp[j];
#pragma unroll
                for (int e = 0; e < 4; ++e)
                    vt[(j * 4 + e) * NPAD + c] = (bf16)v[e];
            }
        } else {
            v8bf z = {};
#pragma unroll
            for (int j = 0; j < 8; ++j) *(v8bf*)(kn + c * HD + j * 8) = z;
#pragma unroll
            for (int d = 0; d < HD; ++d) vt[d * NPAD + c] = (bf16)0.f;
        }
    }
    __syncthreads();

    // ---- A fragments for q (two K-chunks of 32 over d) ----
    const bf16* qrow = qn + l16 * HD;
    v16bf aq0 = make_frag(*(const v8bf*)(qrow + ha),
                          *(const v8bf*)(qrow + ha + 16));
    v16bf aq1 = make_frag(*(const v8bf*)(qrow + 32 + ha),
                          *(const v8bf*)(qrow + 32 + ha + 16));

    // ---- scores: qn . kn^T  (10 N-tiles x 2 K-chunks) ----
    const int rb = (lane >> 4) ? 8 : 0;
#pragma unroll
    for (int ntile = 0; ntile < NPAD / 16; ++ntile) {
        const int coln = ntile * 16 + l16;
        const bf16* kcol = kn + coln * HD;
        v16bf b0 = make_frag(*(const v8bf*)(kcol + ha),
                             *(const v8bf*)(kcol + ha + 16));
        v16bf b1 = make_frag(*(const v8bf*)(kcol + 32 + ha),
                             *(const v8bf*)(kcol + 32 + ha + 16));
        v8f c = {};
        c = __builtin_amdgcn_wmma_f32_16x16x32_bf16(false, aq0, false, b0,
                                                    (short)0, c, false, false);
        c = __builtin_amdgcn_wmma_f32_16x16x32_bf16(false, aq1, false, b1,
                                                    (short)0, c, false, false);
#pragma unroll
        for (int r = 0; r < 8; ++r) sc[(rb + r) * NPAD + coln] = c[r];
    }
    __syncthreads();

    // ---- softmax with per-row 128-window mask + distance prior ----
    if (lane < 16) {
        const int t  = t0 + lane;
        const int st = iclamp(t - FW / 2, 0, T_LEN - FW);
        float* srow  = sc + lane * NPAD;
        float mx = -1e30f;
        for (int c = 0; c < UNW + NGLB; ++c) {
            float v;
            if (c < UNW) {
                int ki = lo + c;
                if (ki >= st && ki < st + FW)
                    v = srow[c] + ALPHA * __expf(-fabsf((float)(ki - t)) / TAU_F);
                else
                    v = -1e30f;
            } else {
                v = srow[c];
            }
            srow[c] = v;
            mx = fmaxf(mx, v);
        }
        float sum = 0.f;
        for (int c = 0; c < UNW + NGLB; ++c) {
            float e = __expf(srow[c] - mx);
            sum += e;
            srow[c] = e;
        }
        float inv = 1.f / sum;
#pragma unroll
        for (int c8 = 0; c8 < NPAD / 8; ++c8) {
            v8bf o;
#pragma unroll
            for (int e = 0; e < 8; ++e) {
                int c = c8 * 8 + e;
                o[e] = (bf16)((c < UNW + NGLB) ? srow[c] * inv : 0.f);
            }
            *(v8bf*)(wt + lane * NPAD + c8 * 8) = o;
        }
    }
    __syncthreads();

    // ---- context: wt . V  (4 D-tiles x 5 K-chunks over 160 keys) ----
    const bf16* wrow = wt + l16 * NPAD;
#pragma unroll
    for (int dtile = 0; dtile < HD / 16; ++dtile) {
        const int d = dtile * 16 + l16;
        const bf16* vrow = vt + d * NPAD;
        v8f c = {};
#pragma unroll
        for (int kc = 0; kc < NPAD / 32; ++kc) {
            v16bf aw = make_frag(*(const v8bf*)(wrow + kc * 32 + ha),
                                 *(const v8bf*)(wrow + kc * 32 + ha + 16));
            v16bf bv = make_frag(*(const v8bf*)(vrow + kc * 32 + ha),
                                 *(const v8bf*)(vrow + kc * 32 + ha + 16));
            c = __builtin_amdgcn_wmma_f32_16x16x32_bf16(
                false, aw, false, bv, (short)0, c, false, false);
        }
#pragma unroll
        for (int r = 0; r < 8; ++r)
            Out[(t0 + rb + r) * HID + h * HD + d] = c[r];
    }
}

// ---------------------------------------------------------------------------
// Host launcher
// ---------------------------------------------------------------------------
extern "C" void kernel_launch(void* const* d_in, const int* in_sizes, int n_in,
                              void* d_out, int out_size, void* d_ws, size_t ws_size,
                              hipStream_t stream) {
    const float* hs = (const float*)d_in[0];
    const float* Wq = (const float*)d_in[1];
    const float* bq = (const float*)d_in[2];
    const float* Wk = (const float*)d_in[3];
    const float* bk = (const float*)d_in[4];
    const float* Wv = (const float*)d_in[5];
    const float* bv = (const float*)d_in[6];
    float* out = (float*)d_out;

    char* ws = (char*)d_ws;
    size_t off = 0;
    auto carve = [&](size_t bytes) -> void* {
        void* p = ws + off;
        off = (off + bytes + 255) & ~(size_t)255;
        return p;
    };
    bf16*  Xb   = (bf16*)carve((size_t)T_LEN * HID * sizeof(bf16));
    bf16*  Wqb  = (bf16*)carve((size_t)HID * HID * sizeof(bf16));
    bf16*  Wkb  = (bf16*)carve((size_t)HID * HID * sizeof(bf16));
    bf16*  Wvb  = (bf16*)carve((size_t)HID * HID * sizeof(bf16));
    float* Qh   = (float*)carve((size_t)NHEAD * T_LEN * HD * sizeof(float));
    float* Kh   = (float*)carve((size_t)NHEAD * T_LEN * HD * sizeof(float));
    float* Vh   = (float*)carve((size_t)NHEAD * T_LEN * HD * sizeof(float));
    int*   gidx = (int*)carve((size_t)NHEAD * NGLB * sizeof(int));

    // 1) convert to bf16 (8 elems/thread)
    convert_bf16_kernel<<<(T_LEN * HID / 8 + 255) / 256, 256, 0, stream>>>(
        hs, Wq, Wk, Wv, Xb, Wqb, Wkb, Wvb);

    // 2) QKV projections (WMMA bf16, f32 accumulate)
    dim3 ggrid(T_LEN / 16, HID / 16 / 8);
    qkv_gemm_kernel<<<ggrid, 256, 0, stream>>>(Xb, Wqb, bq, Qh);
    qkv_gemm_kernel<<<ggrid, 256, 0, stream>>>(Xb, Wkb, bk, Kh);
    qkv_gemm_kernel<<<ggrid, 256, 0, stream>>>(Xb, Wvb, bv, Vh);

    // 3) global key selection
    stats_kernel<<<NHEAD, 256, 0, stream>>>(Kh, gidx);

    // 4) windowed + global attention (WMMA)
    attn_kernel<<<dim3(T_LEN / 16, NHEAD), 32, 0, stream>>>(Qh, Kh, Vh, gidx, out);
}
